// ECCLayer_35742717838041
// MI455X (gfx1250) — matrix operand order repaired and checked by reading, not compiled, not run on hardware
//
#include <hip/hip_runtime.h>

#define BN_EPS 1e-5f

typedef __attribute__((ext_vector_type(2)))  _Float16 h2;
typedef __attribute__((ext_vector_type(16))) _Float16 v16h;
typedef __attribute__((ext_vector_type(8)))  float    v8f;

union AFrag { v16h v; h2 p[8]; unsigned u[8]; };
union BFrag { v16h v; unsigned u[8]; };

__device__ inline unsigned pkh(float a, float b) {
  unsigned short lo = __builtin_bit_cast(unsigned short, (_Float16)a);
  unsigned short hi = __builtin_bit_cast(unsigned short, (_Float16)b);
  return (unsigned)lo | ((unsigned)hi << 16);
}

// ---------------------------------------------------------------------------
// Zero agg [N*16] and stats [32]
// ---------------------------------------------------------------------------
__global__ __launch_bounds__(256) void ecc_zero_kernel(float* __restrict__ agg,
                                                       float* __restrict__ stat,
                                                       int n) {
  int i = blockIdx.x * 256 + threadIdx.x;
  if (i < n) agg[i] = 0.f;
  if (i < 32) stat[i] = 0.f;
}

// ---------------------------------------------------------------------------
// Edge kernel, K=272 formulation:
//   msg[e,o] = sum_{kflat<256} z[e,kflat] * w2rr[kflat,o]   (z = h (x) xs)
//            + sum_{i} xs[e,i] * b2[i*16+o]                 (rows 256..271)
// 9 chained v_wmma_f32_16x16x32_f16 per 16-edge tile; the accumulator IS msg.
// All 9 A fragments are pre-built (no VALU writes inside the WMMA chain ->
// no WMMA->VALU WAR NOPs); B fragments are double-buffered so each step's
// LDS latency hides under the previous WMMA.
// ---------------------------------------------------------------------------
__global__ __launch_bounds__(256) void ecc_edge_kernel(
    const float* __restrict__ x,      // [N,16]
    const int*   __restrict__ src,    // [E]
    const int*   __restrict__ dst,    // [E]
    const float* __restrict__ attr,   // [E]
    const float* __restrict__ w1,     // [16]
    const float* __restrict__ b1,     // [16]
    const float* __restrict__ w2,     // [16,256]
    const float* __restrict__ b2,     // [256]
    float*       __restrict__ agg,    // [N,16] (pre-zeroed)
    int E_, int n_tiles)
{
  // Pre-packed B fragments of w2rr (+ b2 rows), f16 pairs along K.
  // Entry [st*256 + j*32 + lane]: rows (32*st + (j>=4?16:0) + (lane>=16?8:0) + 2(j&3), +1),
  // column o = lane&15.
  __shared__ unsigned lds_wb[9 * 256];     // 9 KB
  __shared__ int      lds_dst[8][16];

  const int tid = threadIdx.x;
  for (int qq = tid; qq < 9 * 256; qq += 256) {
    int st = qq >> 8, rem = qq & 255, j = (rem >> 5) & 7, l32 = rem & 31;
    int o = l32 & 15;
    int row = 32 * st + ((j >= 4) ? 16 : 0) + ((l32 >= 16) ? 8 : 0) + 2 * (j & 3);
    float v0, v1;
    if (row < 256) {        // w2rr rows: kflat = k*16+i -> w2[k, i*16+o]
      int k0 = row >> 4, i0 = row & 15;
      int k1 = (row + 1) >> 4, i1 = (row + 1) & 15;
      v0 = w2[k0 * 256 + i0 * 16 + o];
      v1 = w2[k1 * 256 + i1 * 16 + o];
    } else if (row < 272) { // bias rows: b2[i*16+o]
      v0 = b2[(row - 256) * 16 + o];
      v1 = b2[(row - 255) * 16 + o];
    } else {
      v0 = 0.f; v1 = 0.f;
    }
    lds_wb[st * 256 + j * 32 + l32] = pkh(v0, v1);
  }
  __syncthreads();

  const int lane  = tid & 31;
  const int wave  = tid >> 5;
  const int hsel  = lane >> 4;      // lane half: i-slice 0..7 vs 8..15
  const int lm    = lane & 15;      // edge row within tile / output column o

  // uniform filter-net weights (scalar loads)
  float w1v[16], b1v[16];
  #pragma unroll
  for (int k = 0; k < 16; ++k) { w1v[k] = w1[k]; b1v[k] = b1[k]; }

  const int wstep = gridDim.x * 8;
  for (int tile = blockIdx.x * 8 + wave; tile < n_tiles; tile += wstep) {
    const int  tb   = tile * 16;
    const int  e_st = tb + lm;
    const bool stv  = e_st < E_;

    // prefetch next tile's edge-stream data (global_prefetch_b8, speculative)
    {
      int ne = (tile + wstep) * 16 + lm;
      if (ne < E_) {
        __builtin_prefetch(&src[ne], 0, 1);
        __builtin_prefetch(&attr[ne], 0, 1);
      }
    }

    const int sidx = stv ? src[e_st] : 0;
    if (hsel == 0) lds_dst[wave][lm] = stv ? dst[e_st] : 0;

    // this lane's x[src] slice: row = its edge (lm), i-range = its half
    const float4* xv = (const float4*)x;
    float4 p = stv ? xv[sidx * 4 + hsel * 2]     : make_float4(0.f, 0.f, 0.f, 0.f);
    float4 q = stv ? xv[sidx * 4 + hsel * 2 + 1] : make_float4(0.f, 0.f, 0.f, 0.f);
    h2 xsh[4];
    xsh[0] = h2{(_Float16)p.x, (_Float16)p.y};
    xsh[1] = h2{(_Float16)p.z, (_Float16)p.w};
    xsh[2] = h2{(_Float16)q.x, (_Float16)q.y};
    xsh[3] = h2{(_Float16)q.z, (_Float16)q.w};

    // filter-net hidden h[k] = relu(attr*w1+b1), packed pairs (v_cvt_pk)
    const float attrv = stv ? attr[e_st] : 0.f;
    h2 hpk[8];
    #pragma unroll
    for (int s2 = 0; s2 < 8; ++s2) {
      float h0 = fmaxf(attrv * w1v[2 * s2]     + b1v[2 * s2],     0.f);
      float h1 = fmaxf(attrv * w1v[2 * s2 + 1] + b1v[2 * s2 + 1], 0.f);
      hpk[s2] = h2{(_Float16)h0, (_Float16)h1};
    }

    // Pre-build ALL 9 A fragments: z outer-product slices + bias slice.
    // (no VALU writes inside the WMMA chain below)
    AFrag afs[9];
    #pragma unroll
    for (int st = 0; st < 8; ++st) {
      h2 hlo = h2{hpk[st][0], hpk[st][0]};   // op_sel broadcast
      h2 hhi = h2{hpk[st][1], hpk[st][1]};
      #pragma unroll
      for (int j = 0; j < 4; ++j) {
        afs[st].p[j]     = hlo * xsh[j];     // v_pk_mul_f16
        afs[st].p[j + 4] = hhi * xsh[j];
      }
    }
    #pragma unroll
    for (int j = 0; j < 4; ++j) {
      afs[8].p[j]     = xsh[j];
      afs[8].p[j + 4] = h2{(_Float16)0.f, (_Float16)0.f};
    }

    // WMMA chain with double-buffered B fragments
    v8f acc = {0.f, 0.f, 0.f, 0.f, 0.f, 0.f, 0.f, 0.f};
    BFrag bcur, bnxt;
    #pragma unroll
    for (int j = 0; j < 8; ++j) bcur.u[j] = lds_wb[j * 32 + lane];
    #pragma unroll
    for (int st = 0; st < 9; ++st) {
      if (st < 8) {
        #pragma unroll
        for (int j = 0; j < 8; ++j)
          bnxt.u[j] = lds_wb[(st + 1) * 256 + j * 32 + lane];
      }
      acc = __builtin_amdgcn_wmma_f32_16x16x32_f16(
          false, afs[st].v, false, bcur.v, (short)0, acc, false, false);
      if (st < 8) bcur = bnxt;
    }

    // cross-lane LDS visibility for lds_dst (wave-internal)
    asm volatile("s_wait_dscnt 0x0" ::: "memory");

    // acc == msg: VGPR r holds edge (r + hsel*8), column o = lm
    #pragma unroll
    for (int r = 0; r < 8; ++r) {
      int er = r + hsel * 8;
      if (tb + er < E_) {
        int dn = lds_dst[wave][er];
        (void)__hip_atomic_fetch_add(&agg[dn * 16 + lm], acc[r],
                                     __ATOMIC_RELAXED, __HIP_MEMORY_SCOPE_AGENT);
      }
    }
  }
}

// ---------------------------------------------------------------------------
// Node kernel: pre = relu(agg + x@root + bias) via WMMA (C = agg; bias folded
// into the K=16 row of the root B-fragment with A[m,16]=1). Accumulates
// per-column sum / sumsq for BatchNorm. pre may alias agg.
// ---------------------------------------------------------------------------
__global__ __launch_bounds__(256) void ecc_node_kernel(
    const float* __restrict__ x,     // [N,16]
    const float* __restrict__ agg,   // [N,16]
    const float* __restrict__ root,  // [16,16]
    const float* __restrict__ bias,  // [16]
    float*       __restrict__ pre,   // [N,16]
    float*       __restrict__ stat,  // [32] sums / sumsq
    int Nn, int n_tiles)
{
  __shared__ float red[2][256];
  const int tid  = threadIdx.x;
  const int lane = tid & 31, wave = tid >> 5;
  const int hsel = lane >> 4, lm = lane & 15, kbase = hsel * 8;

  // B fragment = root (K padded 16->32); K=16 row carries bias; hoisted.
  BFrag bf;
  #pragma unroll
  for (int j = 0; j < 8; ++j)
    bf.u[j] = (j < 4)
                  ? pkh(root[(kbase + 2 * j) * 16 + lm], root[(kbase + 2 * j + 1) * 16 + lm])
                  : ((j == 4 && hsel == 0) ? pkh(bias[lm], 0.f) : 0u);

  float s1 = 0.f, s2 = 0.f;

  const int wstep = gridDim.x * 8;
  for (int tile = blockIdx.x * 8 + wave; tile < n_tiles; tile += wstep) {
    const int nb = tile * 16;
    const int m  = nb + lm;
    const bool mv = m < Nn;

    const float4* xv = (const float4*)x;
    float4 p = mv ? xv[m * 4 + hsel * 2]     : make_float4(0.f, 0.f, 0.f, 0.f);
    float4 q = mv ? xv[m * 4 + hsel * 2 + 1] : make_float4(0.f, 0.f, 0.f, 0.f);
    v16h af;
    af[0] = (_Float16)p.x; af[1] = (_Float16)p.y;
    af[2] = (_Float16)p.z; af[3] = (_Float16)p.w;
    af[4] = (_Float16)q.x; af[5] = (_Float16)q.y;
    af[6] = (_Float16)q.z; af[7] = (_Float16)q.w;
    #pragma unroll
    for (int j = 8; j < 16; ++j) af[j] = (_Float16)0.f;
    af[8] = (hsel == 0) ? (_Float16)1.f : (_Float16)0.f;   // bias row

    v8f c;
    #pragma unroll
    for (int r = 0; r < 8; ++r) {
      int node = nb + r + kbase;
      c[r] = (node < Nn) ? agg[node * 16 + lm] : 0.f;
    }

    v8f d = __builtin_amdgcn_wmma_f32_16x16x32_f16(
        false, af, false, bf.v, (short)0, c, false, false);

    #pragma unroll
    for (int r = 0; r < 8; ++r) {
      int node = nb + r + kbase;
      if (node < Nn) {
        float o = fmaxf(d[r], 0.f);
        pre[node * 16 + lm] = o;
        s1 += o;
        s2 += o * o;
      }
    }
  }

  red[0][tid] = s1;
  red[1][tid] = s2;
  __syncthreads();
  if (tid < 16) {
    float a1 = 0.f, a2 = 0.f;
    for (int j = tid; j < 256; j += 16) { a1 += red[0][j]; a2 += red[1][j]; }
    (void)__hip_atomic_fetch_add(&stat[tid],      a1, __ATOMIC_RELAXED, __HIP_MEMORY_SCOPE_AGENT);
    (void)__hip_atomic_fetch_add(&stat[16 + tid], a2, __ATOMIC_RELAXED, __HIP_MEMORY_SCOPE_AGENT);
  }
}

// ---------------------------------------------------------------------------
// BatchNorm apply: out = (pre - mean) * rsqrt(var + eps) * gamma + beta
// ---------------------------------------------------------------------------
__global__ __launch_bounds__(256) void ecc_bn_kernel(
    const float* __restrict__ pre, const float* __restrict__ stat,
    const float* __restrict__ gamma, const float* __restrict__ beta,
    float* __restrict__ out, int Nn)
{
  int i = blockIdx.x * 256 + threadIdx.x;
  if (i >= Nn * 16) return;
  int o = i & 15;
  float invN = 1.f / (float)Nn;
  float mean = stat[o] * invN;
  float var  = stat[16 + o] * invN - mean * mean;
  out[i] = (pre[i] - mean) * rsqrtf(var + BN_EPS) * gamma[o] + beta[o];
}

// ---------------------------------------------------------------------------
// Host launcher: 3 layers, each = zero -> edge -> node -> bn
// ---------------------------------------------------------------------------
extern "C" void kernel_launch(void* const* d_in, const int* in_sizes, int n_in,
                              void* d_out, int out_size, void* d_ws, size_t ws_size,
                              hipStream_t stream) {
  const float* x0    = (const float*)d_in[0];
  const int*   ei    = (const int*)d_in[1];
  const float* eattr = (const float*)d_in[2];

  const int E_ = in_sizes[2];          // edge_attr is [E,1]
  const int Nn = in_sizes[0] / 16;     // x is [N,16]
  const int* srcp = ei;
  const int* dstp = ei + E_;

  float* ws   = (float*)d_ws;
  float* agg  = ws;                    // [N*16]  (also reused as `pre` in-place)
  float* xbA  = agg + (size_t)Nn * 16; // [N*16]
  float* xbB  = xbA + (size_t)Nn * 16; // [N*16]
  float* stat = xbB + (size_t)Nn * 16; // [32]

  const int n_et = (E_ + 15) / 16;
  const int n_nt = (Nn + 15) / 16;
  const int edge_blocks = (n_et + 7) / 8;
  const int node_blocks = (n_nt + 7) / 8;
  const int elem_blocks = (Nn * 16 + 255) / 256;

  const float* xin = x0;
  for (int L = 0; L < 3; ++L) {
    const int base = 3 + L * 8;
    const float* w1    = (const float*)d_in[base + 0];
    const float* b1    = (const float*)d_in[base + 1];
    const float* w2    = (const float*)d_in[base + 2];
    const float* b2    = (const float*)d_in[base + 3];
    const float* root  = (const float*)d_in[base + 4];
    const float* bias  = (const float*)d_in[base + 5];
    const float* gamma = (const float*)d_in[base + 6];
    const float* beta  = (const float*)d_in[base + 7];

    float* xout = (L == 2) ? (float*)d_out : ((L == 0) ? xbA : xbB);

    ecc_zero_kernel<<<elem_blocks, 256, 0, stream>>>(agg, stat, Nn * 16);
    ecc_edge_kernel<<<edge_blocks, 256, 0, stream>>>(
        xin, srcp, dstp, eattr, w1, b1, w2, b2, agg, E_, n_et);
    ecc_node_kernel<<<node_blocks, 256, 0, stream>>>(
        xin, agg, root, bias, /*pre in-place=*/agg, stat, Nn, n_nt);
    ecc_bn_kernel<<<elem_blocks, 256, 0, stream>>>(
        agg, stat, gamma, beta, xout, Nn);

    xin = xout;
  }
}
